// GCNLayer_LiDAR_36206574305748
// MI455X (gfx1250) — matrix operand order repaired and checked by reading, not compiled, not run on hardware
//
#include <hip/hip_runtime.h>
#include <hip/hip_bf16.h>

typedef __attribute__((ext_vector_type(16))) _Float16 v16h;
typedef __attribute__((ext_vector_type(8)))  _Float16 v8h;
typedef __attribute__((ext_vector_type(8)))  float    v8f;

#define N_NODES 8192
#define D_IN    128
#define D_OUT   128
#define D_TH    256

// ---------------------------------------------------------------------------
// WMMA helpers (CDNA5 wave32, v_wmma_f32_16x16x32_f16)
// ---------------------------------------------------------------------------
__device__ __forceinline__ v8f wmma_f16(v16h a, v16h b, v8f c) {
    // 8 args: (neg_a, A, neg_b, B, c_mod, C, reuse_a, reuse_b)
    return __builtin_amdgcn_wmma_f32_16x16x32_f16(false, a, false, b,
                                                  (short)0, c, false, false);
}

// A operand (16x32 f16): lane L -> row M=L&15, 16 contiguous K-halves kb..kb+15,
// kb = (L>>4)*16. Source is row-major f16 with leading dim `lda`.
__device__ __forceinline__ v16h load_a16(const _Float16* __restrict__ base, int lda,
                                         int row0, int kk, int lane) {
    int r  = row0 + (lane & 15);
    int ko = (lane >> 4) << 4;                  // 0 or 16
    const _Float16* p = base + (size_t)r * lda + kk + ko;
    v8h lo = *(const v8h*)p;
    v8h hi = *(const v8h*)(p + 8);
    v16h a;
#pragma unroll
    for (int t = 0; t < 8; ++t) { a[t] = lo[t]; a[t + 8] = hi[t]; }
    return a;
}

// B operand (32x16 f16): lane L -> column N=L&15, 16 contiguous K-halves at
// offset (L>>4)*16. B[k][n] = src[col0+n][k] (i.e. B = src^T), so each lane
// reads a contiguous row chunk of `src`.
__device__ __forceinline__ v16h load_b16(const _Float16* __restrict__ base, int lda,
                                         int col0, int kk, int lane) {
    int c  = col0 + (lane & 15);
    int ko = (lane >> 4) << 4;                  // 0 or 16
    const _Float16* p = base + (size_t)c * lda + kk + ko;
    return *(const v16h*)p;
}

// ---------------------------------------------------------------------------
// 1) BatchNorm column stats (mean, rsqrt(var+eps)) over 8192 rows x 128 cols
// ---------------------------------------------------------------------------
__global__ void k_bn_stats(const float* __restrict__ H,
                           float* __restrict__ mean, float* __restrict__ rstd) {
    __shared__ float s1[256], s2[256];
    int c = blockIdx.x;                          // one column per block
    float s = 0.f, q = 0.f;
    for (int r = threadIdx.x; r < N_NODES; r += 256) {
        float v = H[(size_t)r * D_IN + c];
        s += v; q += v * v;
    }
    s1[threadIdx.x] = s; s2[threadIdx.x] = q; __syncthreads();
    for (int o = 128; o > 0; o >>= 1) {
        if (threadIdx.x < o) {
            s1[threadIdx.x] += s1[threadIdx.x + o];
            s2[threadIdx.x] += s2[threadIdx.x + o];
        }
        __syncthreads();
    }
    if (threadIdx.x == 0) {
        float m   = s1[0] * (1.0f / N_NODES);
        float var = s2[0] * (1.0f / N_NODES) - m * m;
        mean[c] = m;
        rstd[c] = rsqrtf(var + 1e-5f);
    }
}

// ---------------------------------------------------------------------------
// 2) BatchNorm apply: Hn (f32) + f16 copy for WMMA operands
// ---------------------------------------------------------------------------
__global__ void k_bn_apply(const float* __restrict__ H,
                           const float* __restrict__ mean, const float* __restrict__ rstd,
                           const float* __restrict__ gamma, const float* __restrict__ beta,
                           float* __restrict__ Hn, _Float16* __restrict__ Hn_h) {
    int g = blockIdx.x * 256 + threadIdx.x;
    if (g >= N_NODES * D_IN) return;
    int c = g & (D_IN - 1);
    float v = (H[g] - mean[c]) * rstd[c] * gamma[c] + beta[c];
    Hn[g]   = v;
    Hn_h[g] = (_Float16)v;
}

// ---------------------------------------------------------------------------
// 3) f32 -> f16 cast (weights)
// ---------------------------------------------------------------------------
__global__ void k_cast_f16(const float* __restrict__ src, _Float16* __restrict__ dst, int n) {
    int g = blockIdx.x * 256 + threadIdx.x;
    if (g < n) dst[g] = (_Float16)src[g];
}

// ---------------------------------------------------------------------------
// 4) Hx = Hn @ W_theta^T + b_theta   [8192 x 256], stored f16
//    one wave per 16x16 tile, K = 128 (4 WMMAs)
// ---------------------------------------------------------------------------
__global__ void k_gemm_theta(const _Float16* __restrict__ Hn_h,
                             const _Float16* __restrict__ Wth_h,
                             const float* __restrict__ b_theta,
                             _Float16* __restrict__ Hx_h) {
    int lane = threadIdx.x & 31;
    int wave = (blockIdx.x * blockDim.x + threadIdx.x) >> 5;
    int mt = wave >> 4;                          // 512 row tiles
    int nt = wave & 15;                          // 16 col tiles
    int row0 = mt * 16, col0 = nt * 16;
    v8f acc = {};
#pragma unroll
    for (int kk = 0; kk < D_IN; kk += 32) {
        v16h a = load_a16(Hn_h, D_IN, row0, kk, lane);
        v16h b = load_b16(Wth_h, D_IN, col0, kk, lane);
        acc = wmma_f16(a, b, acc);
    }
    int mb = (lane >> 4) << 3;
    int n  = lane & 15;
    int col = col0 + n;
    float bt = b_theta[col];
#pragma unroll
    for (int v = 0; v < 8; ++v) {
        int i = row0 + mb + v;
        Hx_h[(size_t)i * D_TH + col] = (_Float16)(acc[v] + bt);
    }
}

// ---------------------------------------------------------------------------
// 5) sq[i] = sum_k Hx[i][k]^2
// ---------------------------------------------------------------------------
__global__ void k_sqsum(const _Float16* __restrict__ Hx_h, float* __restrict__ sq) {
    int i = blockIdx.x * 256 + threadIdx.x;
    if (i >= N_NODES) return;
    const _Float16* p = Hx_h + (size_t)i * D_TH;
    float s = 0.f;
#pragma unroll 8
    for (int k = 0; k < D_TH; ++k) {
        float v = (float)p[k];
        s += v * v;
    }
    sq[i] = s;
}

// ---------------------------------------------------------------------------
// 6) X = Hn @ W_out^T + b_out   [8192 x 128] f32
// ---------------------------------------------------------------------------
__global__ void k_gemm_xout(const _Float16* __restrict__ Hn_h,
                            const _Float16* __restrict__ Wout_h,
                            const float* __restrict__ b_out,
                            float* __restrict__ X) {
    int lane = threadIdx.x & 31;
    int wave = (blockIdx.x * blockDim.x + threadIdx.x) >> 5;
    int mt = wave >> 3;                          // 512 row tiles
    int nt = wave & 7;                           // 8 col tiles
    int row0 = mt * 16, col0 = nt * 16;
    v8f acc = {};
#pragma unroll
    for (int kk = 0; kk < D_IN; kk += 32) {
        v16h a = load_a16(Hn_h, D_IN, row0, kk, lane);
        v16h b = load_b16(Wout_h, D_IN, col0, kk, lane);
        acc = wmma_f16(a, b, acc);
    }
    int mb = (lane >> 4) << 3;
    int n  = lane & 15;
    int col = col0 + n;
    float bo = b_out[col];
#pragma unroll
    for (int v = 0; v < 8; ++v) {
        int i = row0 + mb + v;
        X[(size_t)i * D_OUT + col] = acc[v] + bo;
    }
}

// ---------------------------------------------------------------------------
// 7) Fused adjacency: per 16x16 tile of the NxN map
//    G = Hx Hx^T (WMMA, K=256)  ->  dist -> exp -> sigmoid -> clip
//    -> DropEdge (symmetric hash mask) -> A_adj mask -> +eye -> threshold
//    Writes A3 directly into d_out's second region (single NxN write).
// ---------------------------------------------------------------------------
__device__ __forceinline__ float drop_keep_uniform(int i, int j) {
    unsigned r = (i <= j) ? (unsigned)i : (unsigned)j;  // upper-tri coords
    unsigned c = (i <= j) ? (unsigned)j : (unsigned)i;
    unsigned long long x = ((unsigned long long)r << 13) | c;
    x += 0x9E3779B97F4A7C15ull;                          // splitmix64
    x = (x ^ (x >> 30)) * 0xBF58476D1CE4E5B9ull;
    x = (x ^ (x >> 27)) * 0x94D049BB133111EBull;
    x ^= x >> 31;
    return (float)(unsigned)(x >> 40) * (1.0f / 16777216.0f);
}

__global__ void k_adj(const _Float16* __restrict__ Hx_h,
                      const float* __restrict__ sq,
                      const float* __restrict__ A_adj,
                      float* __restrict__ A3) {
    int lane = threadIdx.x & 31;
    int wave = (blockIdx.x * blockDim.x + threadIdx.x) >> 5;
    int mt = wave >> 9;                          // 512 row tiles
    int nt = wave & 511;                         // 512 col tiles
    int i0 = mt * 16, j0 = nt * 16;

    // prefetch the A_adj tile rows this wave will consume in its epilogue
    {
        int mb = (lane >> 4) << 3;
        int n  = lane & 15;
#pragma unroll
        for (int v = 0; v < 8; ++v) {
            __builtin_prefetch(&A_adj[(size_t)(i0 + mb + v) * N_NODES + j0 + n], 0, 0);
        }
    }

    v8f acc = {};
#pragma unroll
    for (int kk = 0; kk < D_TH; kk += 32) {
        v16h a = load_a16(Hx_h, D_TH, i0, kk, lane);
        v16h b = load_b16(Hx_h, D_TH, j0, kk, lane);   // B = Hx^T tile
        acc = wmma_f16(a, b, acc);
    }
    int mb = (lane >> 4) << 3;
    int n  = lane & 15;
    int j  = j0 + n;
    float sqj = sq[j];
#pragma unroll
    for (int v = 0; v < 8; ++v) {
        int i = i0 + mb + v;
        float dist = sq[i] + sqj - 2.0f * acc[v];
        dist = fmaxf(dist, 1e-12f);
        float d  = sqrtf(dist);
        float av = __expf(-0.2f * d);
        av = 1.0f / (1.0f + __expf(-av));        // sigmoid
        av = fmaxf(av, 0.1f);                    // clip
        float keep = (drop_keep_uniform(i, j) < 0.8f) ? 1.0f : 0.0f;
        float msk  = (A_adj[(size_t)i * N_NODES + j] > 0.0f) ? 1.0f : 0.0f;
        float a2 = av * keep * msk + ((i == j) ? 1.0f : 0.0f);
        float a3 = (a2 > 0.6f) ? a2 : 0.0f;
        A3[(size_t)i * N_NODES + j] = a3;
    }
}

// ---------------------------------------------------------------------------
// 8) dinv[i] = rsqrt(rowsum(A3[i,:]))   (diag >= 1 so always valid)
// ---------------------------------------------------------------------------
__global__ void k_deg(const float* __restrict__ A3, float* __restrict__ dinv) {
    __shared__ float sh[256];
    int i = blockIdx.x;
    const float* row = A3 + (size_t)i * N_NODES;
    float s = 0.f;
    for (int j = threadIdx.x; j < N_NODES; j += 256) s += row[j];
    sh[threadIdx.x] = s; __syncthreads();
    for (int o = 128; o > 0; o >>= 1) {
        if (threadIdx.x < o) sh[threadIdx.x] += sh[threadIdx.x + o];
        __syncthreads();
    }
    if (threadIdx.x == 0) dinv[i] = rsqrtf(sh[0]);
}

// ---------------------------------------------------------------------------
// 9) XsT[c][j] = f16( X[j][c] * dinv[j] )   -- transposed so final-GEMM B
//    operand loads are contiguous along K.
// ---------------------------------------------------------------------------
__global__ void k_xst(const float* __restrict__ X, const float* __restrict__ dinv,
                      _Float16* __restrict__ XsT) {
    int g = blockIdx.x * 256 + threadIdx.x;      // 128 * 8192
    if (g >= D_OUT * N_NODES) return;
    int c = g >> 13;                             // / 8192
    int j = g & (N_NODES - 1);
    XsT[g] = (_Float16)(X[(size_t)j * D_OUT + c] * dinv[j]);
}

// ---------------------------------------------------------------------------
// 10) out = LeakyReLU( dinv[i] * (A3 @ Xs) )
//     One BLOCK (8 waves) per 16-row strip of A3; wave w owns output
//     column-tile w. All 8 waves stream the same 16x8192 f32 strip, so the
//     redundant A-loads hit WGP$/L2 and HBM traffic stays ~= one A3 pass,
//     while occupancy is 4096 waves (8x the single-wave-per-strip variant)
//     and each wave needs only one v8f accumulator.
// ---------------------------------------------------------------------------
__global__ void k_final(const float* __restrict__ A3,
                        const _Float16* __restrict__ XsT,
                        const float* __restrict__ dinv,
                        float* __restrict__ out) {
    int lane  = threadIdx.x & 31;
    int ntile = threadIdx.x >> 5;                // 0..7 -> output col tile
    int i0    = blockIdx.x * 16;                 // 512 row strips

    int rA  = i0 + (lane & 15);
    int koA = (lane >> 4) << 4;                  // 0 or 16 (f32 elements)
    int koB = (lane >> 4) << 4;                  // 0 or 16 (f16 elements)
    int cB  = ntile * 16 + (lane & 15);

    const float*    rowA = A3  + (size_t)rA * N_NODES + koA;
    const _Float16* rowB = XsT + (size_t)cB * N_NODES + koB;

    v8f acc = {};
    for (int kk = 0; kk < N_NODES; kk += 32) {
        // keep the HBM stream ahead of the consume point (~8 iters = 1KB/row)
        __builtin_prefetch(rowA + kk + 256, 0, 0);

        const float* pa = rowA + kk;
        v16h a;
#pragma unroll
        for (int t = 0; t < 8; ++t) {
            a[t]     = (_Float16)pa[t];
            a[t + 8] = (_Float16)pa[t + 8];
        }
        v16h b = *(const v16h*)(rowB + kk);
        acc = wmma_f16(a, b, acc);
    }

    int mb = (lane >> 4) << 3;
    int n  = lane & 15;
#pragma unroll
    for (int v = 0; v < 8; ++v) {
        int i = i0 + mb + v;
        float val = acc[v] * dinv[i];
        val = (val >= 0.f) ? val : 0.01f * val;              // LeakyReLU(0.01)
        out[(size_t)i * D_OUT + ntile * 16 + n] = val;
    }
}

// ---------------------------------------------------------------------------
// Launch
// ---------------------------------------------------------------------------
static inline size_t align_up(size_t x) { return (x + 255) & ~(size_t)255; }

extern "C" void kernel_launch(void* const* d_in, const int* in_sizes, int n_in,
                              void* d_out, int out_size, void* d_ws, size_t ws_size,
                              hipStream_t stream) {
    (void)in_sizes; (void)n_in; (void)out_size; (void)ws_size;

    const float* H        = (const float*)d_in[0];
    const float* A_adj    = (const float*)d_in[1];
    const float* bn_gamma = (const float*)d_in[2];
    const float* bn_beta  = (const float*)d_in[3];
    const float* W_theta  = (const float*)d_in[4];
    const float* b_theta  = (const float*)d_in[5];
    const float* W_out    = (const float*)d_in[6];
    const float* b_out    = (const float*)d_in[7];

    float* out = (float*)d_out;                                   // [8192 x 128]
    float* A3  = out + (size_t)N_NODES * D_OUT;                   // [8192 x 8192]

    // workspace carve-up
    char* ws = (char*)d_ws;
    float*    mean  = (float*)ws;    ws += align_up(D_IN * sizeof(float));
    float*    rstd  = (float*)ws;    ws += align_up(D_IN * sizeof(float));
    float*    Hn    = (float*)ws;    ws += align_up((size_t)N_NODES * D_IN * sizeof(float));
    _Float16* Hn_h  = (_Float16*)ws; ws += align_up((size_t)N_NODES * D_IN * sizeof(_Float16));
    _Float16* Wth_h = (_Float16*)ws; ws += align_up((size_t)D_TH * D_IN * sizeof(_Float16));
    _Float16* Wout_h= (_Float16*)ws; ws += align_up((size_t)D_OUT * D_IN * sizeof(_Float16));
    _Float16* Hx_h  = (_Float16*)ws; ws += align_up((size_t)N_NODES * D_TH * sizeof(_Float16));
    float*    sq    = (float*)ws;    ws += align_up((size_t)N_NODES * sizeof(float));
    float*    dinv  = (float*)ws;    ws += align_up((size_t)N_NODES * sizeof(float));
    float*    X     = (float*)ws;    ws += align_up((size_t)N_NODES * D_OUT * sizeof(float));
    _Float16* XsT   = (_Float16*)ws; ws += align_up((size_t)D_OUT * N_NODES * sizeof(_Float16));

    // 1) BN stats + apply
    k_bn_stats<<<D_IN, 256, 0, stream>>>(H, mean, rstd);
    k_bn_apply<<<(N_NODES * D_IN + 255) / 256, 256, 0, stream>>>(
        H, mean, rstd, bn_gamma, bn_beta, Hn, Hn_h);

    // 2) weight casts
    k_cast_f16<<<(D_TH * D_IN + 255) / 256, 256, 0, stream>>>(W_theta, Wth_h, D_TH * D_IN);
    k_cast_f16<<<(D_OUT * D_IN + 255) / 256, 256, 0, stream>>>(W_out, Wout_h, D_OUT * D_IN);

    // 3) theta projection (WMMA): 512*16 tiles, 8 waves/block
    k_gemm_theta<<<(512 * 16) / 8, 256, 0, stream>>>(Hn_h, Wth_h, b_theta, Hx_h);
    k_sqsum<<<N_NODES / 256, 256, 0, stream>>>(Hx_h, sq);

    // 4) output projection X (WMMA): 512*8 tiles
    k_gemm_xout<<<(512 * 8) / 8, 256, 0, stream>>>(Hn_h, Wout_h, b_out, X);

    // 5) fused Gram + adjacency pipeline -> A3 (512*512 tiles)
    k_adj<<<(512 * 512) / 8, 256, 0, stream>>>(Hx_h, sq, A_adj, A3);

    // 6) degree normalization
    k_deg<<<N_NODES, 256, 0, stream>>>(A3, dinv);
    k_xst<<<(D_OUT * N_NODES + 255) / 256, 256, 0, stream>>>(X, dinv, XsT);

    // 7) final aggregation GEMM (WMMA): one block (8 waves) per 16-row strip
    k_final<<<512, 256, 0, stream>>>(A3, XsT, dinv, out);
}
